// EncodeProcessDecodeMultiScale_77902116815148
// MI455X (gfx1250) — matrix operand order repaired and checked by reading, not compiled, not run on hardware
//
#include <hip/hip_runtime.h>

// ============================================================================
// MeshGraphNets-style encode-process-decode GNN for MI455X (gfx1250).
//
// Roofline: ~0.3 TFLOP of GEMM work but several GB of gathered activation
// traffic -> bandwidth bound at 23.3 TB/s.  All H=128 GEMMs run on
// v_wmma_f32_16x16x32_f16 (f16 in, f32 accum).  Weights are pre-packed once
// into the exact WMMA B-fragment lane layout so every wave fetches its B tile
// with two coalesced b128 loads that stay resident in the 192MB L2.  Each
// pipeline stage is one fused kernel: gather/concat (vectorized b128 loads)
// -> GEMM1+ReLU -> GEMM2 -> LayerNorm -> (residual | store | atomic scatter).
// ============================================================================

typedef _Float16 half_t;
typedef __attribute__((ext_vector_type(16))) _Float16 v16h;
typedef __attribute__((ext_vector_type(8)))  float    v8f;

#define TM 32            // rows per block (2 row-tiles of 16)
#define NTHREADS 256     // 8 wave32 waves

__device__ __forceinline__ v8f wmma32f16(v16h a, v16h b, v8f c) {
  // D = A(16x32 f16) * B(32x16 f16) + C(16x16 f32)
  return __builtin_amdgcn_wmma_f32_16x16x32_f16(false, a, false, b, (short)0, c,
                                                false, false);
}

// Build the 16x32 f16 A fragment for this lane from an LDS row-major tile.
// ISA 7.12.2 layout: VGPR v holds K = (v>>2)*16 + (lane>=16?8:0) + (v&3)*2, +1
__device__ __forceinline__ v16h load_a_frag(const half_t* __restrict__ base,
                                            int lane) {
  union { v16h v; unsigned int u[8]; } f;
  const int hi8 = (lane >> 4) << 3;
#pragma unroll
  for (int vv = 0; vv < 8; ++vv) {
    const int klo = ((vv >> 2) << 4) + hi8 + ((vv & 3) << 1);
    f.u[vv] = *(const unsigned int*)(base + klo);
  }
  return f.v;
}

// Pack 4 f32 -> 4 f16 and store as one 8-byte LDS write (ds_store_b64).
__device__ __forceinline__ void store4h(half_t* __restrict__ dp, float4 v) {
  union { half_t h[4]; uint2 u; } pk;
  pk.h[0] = (half_t)v.x; pk.h[1] = (half_t)v.y;
  pk.h[2] = (half_t)v.z; pk.h[3] = (half_t)v.w;
  *(uint2*)dp = pk.u;
}

// ----------------------------------------------------------------------------
// Weight pre-pack: W[K][Nout] f32 (row major, y = x @ W) -> f16 fragments.
// Packed element ((ct*nks+ks)*32 + lane)*16 + j  =  W[ks*32 + (j>>3)*16 + (lane>=16?8:0) + (j&7)]
//                                                   [ct*16 + (lane&15)]
// ----------------------------------------------------------------------------
__global__ void pack_w_kernel(const float* __restrict__ w, half_t* __restrict__ out,
                              int K, int Kpad, int Nout) {
  const int total = (Nout / 16) * (Kpad / 32) * 32 * 16;
  int idx = blockIdx.x * blockDim.x + threadIdx.x;
  if (idx >= total) return;
  const int j    = idx & 15;
  const int L    = (idx >> 4) & 31;
  const int rest = idx >> 9;           // ct*nks + ks
  const int nks  = Kpad / 32;
  const int ks   = rest % nks;
  const int ct   = rest / nks;
  const int n = ct * 16 + (L & 15);
  const int k = ks * 32 + ((j >> 3) << 4) + ((L >> 4) & 1) * 8 + (j & 7);
  float v = (k < K) ? w[(size_t)k * Nout + n] : 0.0f;
  out[idx] = (half_t)v;
}

// ----------------------------------------------------------------------------
// Feature construction (normalized, zero-padded to K=32, f16)
// ----------------------------------------------------------------------------
__global__ void node_feat_kernel(const float* __restrict__ phi,
                                 const float* __restrict__ sw,
                                 const float* __restrict__ swr,
                                 const float* __restrict__ ntype,
                                 const float* __restrict__ matp,
                                 const float* __restrict__ mean,
                                 const float* __restrict__ stdv,
                                 half_t* __restrict__ out, int n) {
  int i = blockIdx.x * blockDim.x + threadIdx.x;
  if (i >= n) return;
  float f[11];
  f[0] = phi[i]; f[1] = sw[i]; f[2] = swr[i];
  f[3] = ntype[i * 4 + 0]; f[4] = ntype[i * 4 + 1];
  f[5] = ntype[i * 4 + 2]; f[6] = ntype[i * 4 + 3];
  f[7] = matp[0]; f[8] = matp[1]; f[9] = matp[2]; f[10] = matp[3];
  half_t* o = out + (size_t)i * 32;
#pragma unroll
  for (int k = 0; k < 11; ++k) o[k] = (half_t)((f[k] - mean[k]) / stdv[k]);
#pragma unroll
  for (int k = 11; k < 32; ++k) o[k] = (half_t)0.0f;
}

__global__ void edge_feat_kernel(const float* __restrict__ mesh,
                                 const float* __restrict__ world,
                                 const float* __restrict__ phi,
                                 const int* __restrict__ sidx,
                                 const int* __restrict__ ridx,
                                 const int* __restrict__ map, int use_map,
                                 const float* __restrict__ mean,
                                 const float* __restrict__ stdv,
                                 half_t* __restrict__ out, int m) {
  int i = blockIdx.x * blockDim.x + threadIdx.x;
  if (i >= m) return;
  int s = sidx[i], r = ridx[i];
  if (use_map) { s = map[s]; r = map[r]; }
  float rx = mesh[s * 2 + 0] - mesh[r * 2 + 0];
  float ry = mesh[s * 2 + 1] - mesh[r * 2 + 1];
  float d  = sqrtf(rx * rx + ry * ry);
  float wx = world[s * 2 + 0] - world[r * 2 + 0];
  float wy = world[s * 2 + 1] - world[r * 2 + 1];
  float dw = sqrtf(wx * wx + wy * wy);
  float rp = phi[s] - phi[r];
  float f[7] = {rx, ry, d, wx, wy, dw, rp};
  half_t* o = out + (size_t)i * 32;
#pragma unroll
  for (int k = 0; k < 7; ++k) o[k] = (half_t)((f[k] - mean[k]) / stdv[k]);
#pragma unroll
  for (int k = 7; k < 32; ++k) o[k] = (half_t)0.0f;
}

__global__ void gather_rows_kernel(const float* __restrict__ src,
                                   const int* __restrict__ idx,
                                   float* __restrict__ dst, int m) {
  int p = blockIdx.x * blockDim.x + threadIdx.x;     // float4 granule
  if (p >= m * 32) return;
  int r = p >> 5, c = (p & 31) << 2;
  *(float4*)(dst + (size_t)r * 128 + c) =
      *(const float4*)(src + (size_t)idx[r] * 128 + c);
}

__global__ void scatter_rows_kernel(const float* __restrict__ src,
                                    const int* __restrict__ idx,
                                    float* __restrict__ dst, int m) {
  int p = blockIdx.x * blockDim.x + threadIdx.x;     // float4 granule
  if (p >= m * 32) return;
  int r = p >> 5, c = (p & 31) << 2;
  *(float4*)(dst + (size_t)idx[r] * 128 + c) =
      *(const float4*)(src + (size_t)r * 128 + c);
}

// ----------------------------------------------------------------------------
// Fused 2-layer MLP + LayerNorm:  y = LN(relu(A@W1+b1)@W2+b2)*g+beta [+res]
// A: either f16 rows [M,32] (a16 path, encoders) or gather/concat of up to 3
//    f32 [*,128] sources with optional row-index arrays (null = identity).
// Output: optional store and/or atomicAdd scatter (segment_sum of messages).
// Block: 256 threads / 8 waves; wave w owns column tile w (16 of 128 cols)
// for both 16-row tiles.  GEMMs on v_wmma_f32_16x16x32_f16.
// ----------------------------------------------------------------------------
__global__ __launch_bounds__(NTHREADS)
void mlp2_kernel(const half_t* __restrict__ a16,
                 const float* __restrict__ s0, const int* __restrict__ i0,
                 const float* __restrict__ s1, const int* __restrict__ i1,
                 const float* __restrict__ s2, const int* __restrict__ i2,
                 int nseg, int K1,
                 const half_t* __restrict__ w1p, const float* __restrict__ b1,
                 const half_t* __restrict__ w2p, const float* __restrict__ b2,
                 const float* __restrict__ gam, const float* __restrict__ bet,
                 const float* __restrict__ residual,
                 float* __restrict__ out_store,
                 float* __restrict__ out_atomic, const int* __restrict__ scat,
                 int M) {
  extern __shared__ char smem[];
  int*    idx_lds = (int*)smem;                              // [3*TM]
  half_t* a_lds   = (half_t*)(smem + 3 * TM * 4);            // [TM][K1+8]
  const int AS = K1 + 8;
  half_t* h_lds = a_lds + TM * AS;                           // [TM][136]
  const int HS = 136;
  float* o_lds = (float*)(h_lds + TM * HS);                  // [TM][132]
  const int OS = 132;

  const int tid  = threadIdx.x;
  const int lane = tid & 31;
  const int wv   = tid >> 5;
  const int row0 = blockIdx.x * TM;
  (void)nseg;

  // stage gather indices (null index array == identity)
  if (tid < 3 * TM) {
    int seg = tid / TM, r = tid % TM;
    const int* ii = (seg == 0) ? i0 : (seg == 1) ? i1 : i2;
    int gr = row0 + r;
    idx_lds[tid] = ii ? ii[gr] : gr;
  }
  __syncthreads();

  // ---- stage A tile into LDS (f16), fully vectorized ----
  if (a16) {
    for (int p = tid; p < TM * 4; p += NTHREADS) {   // uint4 = 8 halves
      int r = p >> 2, c = (p & 3) << 3;
      *(uint4*)(a_lds + r * AS + c) =
          *(const uint4*)(a16 + (size_t)(row0 + r) * 32 + c);
    }
  } else {
    const int K4 = K1 >> 2;                           // float4 granules per row
    for (int p = tid; p < TM * K4; p += NTHREADS) {
      int r = p / K4, k = (p - r * K4) << 2;
      int seg = k >> 7, cc = k & 127;
      const float* sp = (seg == 0) ? s0 : (seg == 1) ? s1 : s2;
      int srow = idx_lds[seg * TM + r];
      float4 v = *(const float4*)(sp + (size_t)srow * 128 + cc);
      store4h(a_lds + r * AS + k, v);
    }
  }
  __syncthreads();

  const int ct = wv;                 // column tile 0..7
  const int mA = lane & 15;
  const int mhi = (lane >> 4) << 3;
  const int n = ct * 16 + (lane & 15);
  const v8f vz = {};

  // ---- GEMM1: [TM,K1] x [K1,128], +b1, ReLU -> h_lds (f16) ----
  {
    const int nks1 = K1 >> 5;
    v8f acc0 = vz, acc1 = vz;
    for (int ks = 0; ks < nks1; ++ks) {
      const half_t* bp = w1p + (((size_t)ct * nks1 + ks) * 32 + lane) * 16;
      __builtin_prefetch(bp + 32 * 16, 0, 1);   // speculative, DEV scope
      v16h b  = *(const v16h*)bp;
      v16h a0 = load_a_frag(a_lds + mA * AS + ks * 32, lane);
      v16h a1 = load_a_frag(a_lds + (mA + 16) * AS + ks * 32, lane);
      acc0 = wmma32f16(a0, b, acc0);
      acc1 = wmma32f16(a1, b, acc1);
    }
    float bv = b1[n];
#pragma unroll
    for (int r = 0; r < 8; ++r) {
      float y0 = acc0[r] + bv; y0 = y0 > 0.f ? y0 : 0.f;
      float y1 = acc1[r] + bv; y1 = y1 > 0.f ? y1 : 0.f;
      h_lds[(r + mhi) * HS + n]      = (half_t)y0;
      h_lds[(16 + r + mhi) * HS + n] = (half_t)y1;
    }
  }
  __syncthreads();

  // ---- GEMM2: [TM,128] x [128,128], +b2 -> o_lds (f32) ----
  {
    const int nks2 = 4;
    v8f acc0 = vz, acc1 = vz;
#pragma unroll
    for (int ks = 0; ks < nks2; ++ks) {
      v16h b  = *(const v16h*)(w2p + (((size_t)ct * nks2 + ks) * 32 + lane) * 16);
      v16h a0 = load_a_frag(h_lds + mA * HS + ks * 32, lane);
      v16h a1 = load_a_frag(h_lds + (mA + 16) * HS + ks * 32, lane);
      acc0 = wmma32f16(a0, b, acc0);
      acc1 = wmma32f16(a1, b, acc1);
    }
    float bv = b2[n];
#pragma unroll
    for (int r = 0; r < 8; ++r) {
      o_lds[(r + mhi) * OS + n]      = acc0[r] + bv;
      o_lds[(16 + r + mhi) * OS + n] = acc1[r] + bv;
    }
  }
  __syncthreads();

  // ---- LayerNorm + scale/shift + residual + store/atomic-scatter ----
  const float g0 = gam[lane], g1 = gam[lane + 32], g2 = gam[lane + 64], g3 = gam[lane + 96];
  const float t0 = bet[lane], t1 = bet[lane + 32], t2 = bet[lane + 64], t3 = bet[lane + 96];
#pragma unroll
  for (int q = 0; q < 4; ++q) {
    const int rloc = wv * 4 + q;
    const int grow = row0 + rloc;
    float v0 = o_lds[rloc * OS + lane];
    float v1 = o_lds[rloc * OS + lane + 32];
    float v2 = o_lds[rloc * OS + lane + 64];
    float v3 = o_lds[rloc * OS + lane + 96];
    float s = v0 + v1 + v2 + v3;
#pragma unroll
    for (int off = 16; off > 0; off >>= 1) s += __shfl_xor(s, off, 32);
    const float mean = s * (1.0f / 128.0f);
    float d0 = v0 - mean, d1 = v1 - mean, d2 = v2 - mean, d3 = v3 - mean;
    float sq = d0 * d0 + d1 * d1 + d2 * d2 + d3 * d3;
#pragma unroll
    for (int off = 16; off > 0; off >>= 1) sq += __shfl_xor(sq, off, 32);
    const float rstd = rsqrtf(sq * (1.0f / 128.0f) + 1e-5f);
    float y0 = d0 * rstd * g0 + t0;
    float y1 = d1 * rstd * g1 + t1;
    float y2 = d2 * rstd * g2 + t2;
    float y3 = d3 * rstd * g3 + t3;
    if (residual) {
      const float* rp = residual + (size_t)grow * 128;
      y0 += rp[lane]; y1 += rp[lane + 32]; y2 += rp[lane + 64]; y3 += rp[lane + 96];
    }
    if (out_store) {
      float* op = out_store + (size_t)grow * 128;
      op[lane] = y0; op[lane + 32] = y1; op[lane + 64] = y2; op[lane + 96] = y3;
    }
    if (out_atomic) {
      float* op = out_atomic + (size_t)scat[grow] * 128;
      atomicAdd(op + lane, y0);       atomicAdd(op + lane + 32, y1);
      atomicAdd(op + lane + 64, y2);  atomicAdd(op + lane + 96, y3);
    }
  }
}

// ----------------------------------------------------------------------------
// Decoder layer 1: concat(x_h, map_back)[N,256] @ W1[256,64] + b1, Swish
// ----------------------------------------------------------------------------
__global__ __launch_bounds__(NTHREADS)
void dec1_kernel(const float* __restrict__ s0, const float* __restrict__ s1,
                 const half_t* __restrict__ w1p, const float* __restrict__ b1,
                 float* __restrict__ out, int M) {
  extern __shared__ char smem[];
  half_t* a_lds = (half_t*)smem;
  const int K1 = 256, AS = K1 + 8;
  const int tid = threadIdx.x, lane = tid & 31, wv = tid >> 5;
  const int row0 = blockIdx.x * TM;
  for (int p = tid; p < TM * 64; p += NTHREADS) {    // float4 granules
    int r = p >> 6, k = (p & 63) << 2;
    const float* sp = (k < 128) ? s0 : s1;
    float4 v = *(const float4*)(sp + (size_t)(row0 + r) * 128 + (k & 127));
    store4h(a_lds + r * AS + k, v);
  }
  __syncthreads();
  const int rt = wv >> 2, ct = wv & 3;   // 2 row-tiles x 4 col-tiles (Nout=64)
  const int nks = 8;
  v8f acc = {};
  const int mA = (lane & 15) + rt * 16;
  for (int ks = 0; ks < nks; ++ks) {
    v16h b = *(const v16h*)(w1p + (((size_t)ct * nks + ks) * 32 + lane) * 16);
    v16h a = load_a_frag(a_lds + mA * AS + ks * 32, lane);
    acc = wmma32f16(a, b, acc);
  }
  const int n = ct * 16 + (lane & 15);
  const float bv = b1[n];
  const int mhi = (lane >> 4) << 3;
#pragma unroll
  for (int r = 0; r < 8; ++r) {
    float y = acc[r] + bv;
    y = y / (1.0f + __expf(-y));       // Swish(beta=1)
    out[(size_t)(row0 + rt * 16 + r + mhi) * 64 + n] = y;
  }
}

// Decoder layer 2: [N,64] @ W2[64,3] + b2  (tiny; VALU)
__global__ void dec2_kernel(const float* __restrict__ h, const float* __restrict__ w2,
                            const float* __restrict__ b2, float* __restrict__ out,
                            int M) {
  int i = blockIdx.x * blockDim.x + threadIdx.x;
  if (i >= M) return;
  float o0 = b2[0], o1 = b2[1], o2 = b2[2];
  const float* hp = h + (size_t)i * 64;
#pragma unroll 8
  for (int k = 0; k < 64; ++k) {
    float hv = hp[k];
    o0 += hv * w2[k * 3 + 0];
    o1 += hv * w2[k * 3 + 1];
    o2 += hv * w2[k * 3 + 2];
  }
  out[i * 3 + 0] = o0; out[i * 3 + 1] = o1; out[i * 3 + 2] = o2;
}

// ============================================================================
// Host orchestration
// ============================================================================
static inline size_t alignUp(size_t x, size_t a) { return (x + a - 1) & ~(a - 1); }

extern "C" void kernel_launch(void* const* d_in, const int* in_sizes, int n_in,
                              void* d_out, int out_size, void* d_ws, size_t ws_size,
                              hipStream_t stream) {
  (void)in_sizes; (void)n_in; (void)out_size; (void)ws_size;
  const int N = 40000, E = 320000, NC = 20000, EC = 60000;

  const float* world = (const float*)d_in[0];
  const float* mesh  = (const float*)d_in[1];
  const float* phi   = (const float*)d_in[2];
  const float* sw    = (const float*)d_in[3];
  const float* swr   = (const float*)d_in[4];
  const float* ntype = (const float*)d_in[5];
  const float* matp  = (const float*)d_in[6];
  // params pytree flattened with alphabetical dict keys (JAX convention):
  // coarse_enc{b1,b2,beta,g,w1,w2}=0..5, cproc[i]{edge,node}{...}=6..41,
  // dec{b1,b2,w1,w2}=42..45, edge_enc=46..51, node_enc=52..57,
  // norm{ce_mean,ce_std,edge_mean,edge_std,node_mean,node_std}=58..63,
  // proc[i]{edge,node}{...}=64..99
  auto prm = [&](int k) { return (const float*)d_in[7 + k]; };
  const int* edge_index   = (const int*)d_in[107];
  const int* c_edge_index = (const int*)d_in[108];
  const int* first_idx    = (const int*)d_in[109];
  const int* frow = edge_index,        *fcol = edge_index + E;
  const int* crow = c_edge_index,      *ccol = c_edge_index + EC;

  // ---- workspace carve-up ----
  char* base = (char*)d_ws;
  size_t off = 0;
  auto alloc = [&](size_t bytes) -> void* {
    void* p = base + off; off = alignUp(off + bytes, 256); return p;
  };
  float* xA   = (float*)alloc((size_t)N  * 128 * 4);
  float* xB   = (float*)alloc((size_t)N  * 128 * 4);
  float* eA   = (float*)alloc((size_t)E  * 128 * 4);
  float* eB   = (float*)alloc((size_t)E  * 128 * 4);
  float* cxA  = (float*)alloc((size_t)NC * 128 * 4);
  float* cxB  = (float*)alloc((size_t)NC * 128 * 4);
  float* ceA  = (float*)alloc((size_t)EC * 128 * 4);
  float* ceB  = (float*)alloc((size_t)EC * 128 * 4);
  float* aggr = (float*)alloc((size_t)N  * 128 * 4);
  float* mapb = (float*)alloc((size_t)N  * 128 * 4);
  float* hdec = (float*)alloc((size_t)N  * 64 * 4);
  half_t* nodeF  = (half_t*)alloc((size_t)N  * 32 * 2);
  half_t* edgeF  = (half_t*)alloc((size_t)E  * 32 * 2);
  half_t* cedgeF = (half_t*)alloc((size_t)EC * 32 * 2);

  auto pack = [&](const float* w, int K, int Kpad, int Nout) -> half_t* {
    half_t* dst = (half_t*)alloc((size_t)Kpad * Nout * 2);
    int total = (Nout / 16) * (Kpad / 32) * 32 * 16;
    pack_w_kernel<<<(total + 255) / 256, 256, 0, stream>>>(w, dst, K, Kpad, Nout);
    return dst;
  };

  // packed weights (all fragments live in L2; total ~1.5MB)
  half_t* ne_w1p = pack(prm(56), 11, 32, 128);
  half_t* ne_w2p = pack(prm(57), 128, 128, 128);
  half_t* ee_w1p = pack(prm(50), 7, 32, 128);
  half_t* ee_w2p = pack(prm(51), 128, 128, 128);
  half_t* ce_w1p = pack(prm(4), 7, 32, 128);
  half_t* ce_w2p = pack(prm(5), 128, 128, 128);
  half_t *pe_w1p[3], *pe_w2p[3], *pn_w1p[3], *pn_w2p[3];
  half_t *qe_w1p[3], *qe_w2p[3], *qn_w1p[3], *qn_w2p[3];
  for (int i = 0; i < 3; ++i) {
    int pe = 64 + i * 12, pn = pe + 6;     // proc
    int qe = 6 + i * 12,  qn = qe + 6;     // cproc
    pe_w1p[i] = pack(prm(pe + 4), 384, 384, 128);
    pe_w2p[i] = pack(prm(pe + 5), 128, 128, 128);
    pn_w1p[i] = pack(prm(pn + 4), 256, 256, 128);
    pn_w2p[i] = pack(prm(pn + 5), 128, 128, 128);
    qe_w1p[i] = pack(prm(qe + 4), 384, 384, 128);
    qe_w2p[i] = pack(prm(qe + 5), 128, 128, 128);
    qn_w1p[i] = pack(prm(qn + 4), 256, 256, 128);
    qn_w2p[i] = pack(prm(qn + 5), 128, 128, 128);
  }
  half_t* dec_w1p = pack(prm(44), 256, 256, 64);

  // ---- features ----
  node_feat_kernel<<<(N + 255) / 256, 256, 0, stream>>>(
      phi, sw, swr, ntype, matp, prm(62), prm(63), nodeF, N);
  edge_feat_kernel<<<(E + 255) / 256, 256, 0, stream>>>(
      mesh, world, phi, frow, fcol, nullptr, 0, prm(60), prm(61), edgeF, E);
  edge_feat_kernel<<<(EC + 255) / 256, 256, 0, stream>>>(
      mesh, world, phi, crow, ccol, first_idx, 1, prm(58), prm(59), cedgeF, EC);

  auto mlp2 = [&](const half_t* a16,
                  const float* s0, const int* i0, const float* s1, const int* i1,
                  const float* s2, const int* i2, int nseg, int K1,
                  const half_t* w1p, const float* b1,
                  const half_t* w2p, const float* b2,
                  const float* g, const float* bt, const float* res,
                  float* outS, float* outA, const int* scat, int M) {
    size_t sh = 3 * TM * 4 + (size_t)TM * (K1 + 8) * 2 + TM * 136 * 2 + TM * 132 * 4;
    mlp2_kernel<<<M / TM, NTHREADS, sh, stream>>>(
        a16, s0, i0, s1, i1, s2, i2, nseg, K1, w1p, b1, w2p, b2, g, bt, res,
        outS, outA, scat, M);
  };

  // ---- encoders ----  (mlp leaf order: b1,b2,beta,g,w1,w2 -> g=+3, beta=+2)
  mlp2(nodeF, 0, 0, 0, 0, 0, 0, 0, 32, ne_w1p, prm(52), ne_w2p, prm(53),
       prm(55), prm(54), nullptr, xA, nullptr, nullptr, N);
  mlp2(edgeF, 0, 0, 0, 0, 0, 0, 0, 32, ee_w1p, prm(46), ee_w2p, prm(47),
       prm(49), prm(48), nullptr, eA, nullptr, nullptr, E);
  mlp2(cedgeF, 0, 0, 0, 0, 0, 0, 0, 32, ce_w1p, prm(0), ce_w2p, prm(1),
       prm(3), prm(2), nullptr, ceA, nullptr, nullptr, EC);
  gather_rows_kernel<<<(NC * 32 + 255) / 256, 256, 0, stream>>>(xA, first_idx, cxA, NC);

  // ---- fine processor (3 layers) ----
  for (int i = 0; i < 3; ++i) {
    int pe = 64 + i * 12, pn = pe + 6;
    hipMemsetAsync(aggr, 0, (size_t)N * 128 * 4, stream);
    // msg = edgeMLP(x[col], x[row], e); segment_sum at col
    mlp2(nullptr, xA, fcol, xA, frow, eA, nullptr, 3, 384,
         pe_w1p[i], prm(pe), pe_w2p[i], prm(pe + 1), prm(pe + 3), prm(pe + 2),
         nullptr, nullptr, aggr, fcol, E);
    // x_new = nodeMLP(aggr, x) + x
    mlp2(nullptr, aggr, nullptr, xA, nullptr, nullptr, nullptr, 2, 256,
         pn_w1p[i], prm(pn), pn_w2p[i], prm(pn + 1), prm(pn + 3), prm(pn + 2),
         xA, xB, nullptr, nullptr, N);
    // e_new = edgeMLP(x[row], x[col], e) + e
    mlp2(nullptr, xA, frow, xA, fcol, eA, nullptr, 3, 384,
         pe_w1p[i], prm(pe), pe_w2p[i], prm(pe + 1), prm(pe + 3), prm(pe + 2),
         eA, eB, nullptr, nullptr, E);
    float* t;
    t = xA; xA = xB; xB = t;
    t = eA; eA = eB; eB = t;
  }

  // ---- coarse processor (3 layers) ----
  for (int i = 0; i < 3; ++i) {
    int qe = 6 + i * 12, qn = qe + 6;
    hipMemsetAsync(aggr, 0, (size_t)NC * 128 * 4, stream);
    mlp2(nullptr, cxA, ccol, cxA, crow, ceA, nullptr, 3, 384,
         qe_w1p[i], prm(qe), qe_w2p[i], prm(qe + 1), prm(qe + 3), prm(qe + 2),
         nullptr, nullptr, aggr, ccol, EC);
    mlp2(nullptr, aggr, nullptr, cxA, nullptr, nullptr, nullptr, 2, 256,
         qn_w1p[i], prm(qn), qn_w2p[i], prm(qn + 1), prm(qn + 3), prm(qn + 2),
         cxA, cxB, nullptr, nullptr, NC);
    mlp2(nullptr, cxA, crow, cxA, ccol, ceA, nullptr, 3, 384,
         qe_w1p[i], prm(qe), qe_w2p[i], prm(qe + 1), prm(qe + 3), prm(qe + 2),
         ceA, ceB, nullptr, nullptr, EC);
    float* t;
    t = cxA; cxA = cxB; cxB = t;
    t = ceA; ceA = ceB; ceB = t;
  }

  // ---- decoder ----
  hipMemsetAsync(mapb, 0, (size_t)N * 128 * 4, stream);
  scatter_rows_kernel<<<(NC * 32 + 255) / 256, 256, 0, stream>>>(cxA, first_idx, mapb, NC);
  dec1_kernel<<<N / TM, NTHREADS, (size_t)TM * (256 + 8) * 2, stream>>>(
      xA, mapb, dec_w1p, prm(42), hdec, N);
  dec2_kernel<<<(N + 127) / 128, 128, 0, stream>>>(hdec, prm(45), prm(43),
                                                   (float*)d_out, N);
}
